// MultiHeadAttentionQuantum_65481071403396
// MI455X (gfx1250) — matrix operand order repaired and checked by reading, not compiled, not run on hardware
//
#include <hip/hip_runtime.h>

// ---------- types ----------
typedef __attribute__((ext_vector_type(8)))  unsigned short u16x8;
typedef __attribute__((ext_vector_type(16))) __bf16         bf16x16;
typedef __attribute__((ext_vector_type(8)))  float          f32x8;

union ABFrag { u16x8 u[2]; bf16x16 bf; };

__device__ __forceinline__ unsigned short f32_to_bf16(float f) {
  unsigned int u = __builtin_bit_cast(unsigned int, f);
  u += 0x7FFFu + ((u >> 16) & 1u);   // round-to-nearest-even
  return (unsigned short)(u >> 16);
}

// ---------- CDNA5 async global->LDS copy (ASYNCcnt-tracked) ----------
__device__ __forceinline__ void async_copy_b128(unsigned lds_off, const void* gaddr) {
  asm volatile("global_load_async_to_lds_b128 %0, %1, off"
               :: "v"(lds_off), "v"((unsigned long long)gaddr)
               : "memory");
}
__device__ __forceinline__ void wait_async_le4() {
  asm volatile("s_wait_asynccnt 0x4" ::: "memory");
}
__device__ __forceinline__ void wait_async_0() {
  asm volatile("s_wait_asynccnt 0x0" ::: "memory");
}

// ---------- fp32 -> bf16 conversion ----------
__global__ void cvt_bf16_kernel(const float* __restrict__ in,
                                unsigned short* __restrict__ out, int n) {
  int i = (blockIdx.x * 256 + threadIdx.x) * 4;
  if (i + 3 < n) {
    float4 v = *(const float4*)(in + i);
    out[i + 0] = f32_to_bf16(v.x);
    out[i + 1] = f32_to_bf16(v.y);
    out[i + 2] = f32_to_bf16(v.z);
    out[i + 3] = f32_to_bf16(v.w);
  }
}

// Pack Wo's 64 relevant columns (h*64+j, j<4) into (1024 x 64) bf16, N x K layout.
__global__ void pack_wo_kernel(const float* __restrict__ Wo,
                               unsigned short* __restrict__ wob) {
  int i = blockIdx.x * 256 + threadIdx.x;   // 0 .. 1024*64-1
  int nrow = i >> 6;
  int kk   = i & 63;
  wob[i] = f32_to_bf16(Wo[nrow * 1024 + (kk >> 2) * 64 + (kk & 3)]);
}

// ---------- bf16 tiled transpose: (rows x cols) -> (cols x rows), per batch ----------
__global__ __launch_bounds__(256) void transpose_bf16_kernel(
    const unsigned short* __restrict__ in, unsigned short* __restrict__ out,
    int rows, int cols) {
  __shared__ unsigned short tile[32][33];
  const long base = (long)blockIdx.z * rows * cols;
  const int r0 = blockIdx.y * 32, c0 = blockIdx.x * 32;
#pragma unroll
  for (int i = threadIdx.y; i < 32; i += 8)
    tile[i][threadIdx.x] = in[base + (long)(r0 + i) * cols + c0 + threadIdx.x];
  __syncthreads();
#pragma unroll
  for (int i = threadIdx.y; i < 32; i += 8)
    out[base + (long)(c0 + i) * rows + r0 + threadIdx.x] = tile[threadIdx.x][i];
}

// ---------- WMMA GEMM: C = alpha * A(bf16, MxK) @ B^T, B given (N x K) row-major ----
// Tile 128(M) x 128(N) x 32(K); 256 threads = 8 waves as 4(M) x 2(N);
// wave owns 32x64 of C (2x4 WMMA subtiles). Double-buffered LDS filled by
// global_load_async_to_lds_b128, overlapped with WMMA of the previous tile.
#define GBM 128
#define GBN 128
#define GBK 32
#define SLD 40   // padded LDS row stride (elements); 80B (16B-aligned)

template <bool OUT_BF16>
__global__ __launch_bounds__(256) void gemm_wmma_kernel(
    const unsigned short* __restrict__ A,
    const unsigned short* __restrict__ Bm,
    void* __restrict__ Cv,
    int K, int lda, int ldb, int ldc,
    long sA, long sB, long sC, float alpha) {
  __shared__ unsigned short As[2][GBM * SLD];
  __shared__ unsigned short Bs[2][GBN * SLD];

  const int tid  = threadIdx.x;
  const int lane = tid & 31;
  const int wave = tid >> 5;
  const int half = lane >> 4;     // 0: lanes 0-15, 1: lanes 16-31
  const int lrow = lane & 15;
  const int wm   = wave & 3;      // 4 waves along M
  const int wn   = wave >> 2;     // 2 waves along N

  const int m0 = blockIdx.y * GBM;
  const int n0 = blockIdx.x * GBN;
  const int bz = blockIdx.z;

  A  += sA * bz;
  Bm += sB * bz;
  float*          Cf = (float*)Cv + sC * bz;
  unsigned short* Cb = (unsigned short*)Cv + sC * bz;

  f32x8 acc[2][4];
#pragma unroll
  for (int m = 0; m < 2; m++)
#pragma unroll
    for (int n = 0; n < 4; n++)
#pragma unroll
      for (int r = 0; r < 8; r++) acc[m][n][r] = 0.0f;

  // staging coordinates: thread loads rows (lr, lr+64) x 8 bf16 at column lk
  const int lr = tid >> 2;
  const int lk = (tid & 3) * 8;
  const unsigned asBase = (unsigned)(unsigned long long)(void*)&As[0][0];
  const unsigned bsBase = (unsigned)(unsigned long long)(void*)&Bs[0][0];

  auto issue = [&](int buf, int k0) {
    const unsigned aoff = asBase + (unsigned)(buf * GBM * SLD + lr * SLD + lk) * 2u;
    const unsigned short* ga = &A[(long)(m0 + lr) * lda + k0 + lk];
    async_copy_b128(aoff, ga);
    async_copy_b128(aoff + 64u * SLD * 2u, ga + (long)64 * lda);
    const unsigned boff = bsBase + (unsigned)(buf * GBN * SLD + lr * SLD + lk) * 2u;
    const unsigned short* gb = &Bm[(long)(n0 + lr) * ldb + k0 + lk];
    async_copy_b128(boff, gb);
    async_copy_b128(boff + 64u * SLD * 2u, gb + (long)64 * ldb);
  };

  issue(0, 0);
  const int nk = K / GBK;
  for (int kt = 0; kt < nk; ++kt) {
    const int cur = kt & 1;
    if (kt + 1 < nk) {
      issue(cur ^ 1, (kt + 1) * GBK);  // 4 async ops for the next tile
      wait_async_le4();                // current tile's 4 copies complete
    } else {
      wait_async_0();
    }
    __syncthreads();                   // all waves' copies of `cur` visible

    const unsigned short* as_ = As[cur];
    const unsigned short* bs_ = Bs[cur];

    // A fragments (16x32): elem e<8 -> K=half*8+e ; e>=8 -> K=16+half*8+(e-8)
    ABFrag af[2];
#pragma unroll
    for (int m = 0; m < 2; m++) {
      const int row = wm * 32 + m * 16 + lrow;
      af[m].u[0] = *(const u16x8*)&as_[row * SLD + half * 8];
      af[m].u[1] = *(const u16x8*)&as_[row * SLD + 16 + half * 8];
    }
#pragma unroll
    for (int n = 0; n < 4; n++) {
      // B fragment (32x16): lane col N=lrow, elem e<8 -> K=half*8+e ; e>=8 -> K=16+half*8+(e-8)
      const int row = wn * 64 + n * 16 + lrow;
      ABFrag bf_;
      bf_.u[0] = *(const u16x8*)&bs_[row * SLD + half * 8];
      bf_.u[1] = *(const u16x8*)&bs_[row * SLD + 16 + half * 8];
#pragma unroll
      for (int m = 0; m < 2; m++)
        acc[m][n] = __builtin_amdgcn_wmma_f32_16x16x32_bf16(
            false, af[m].bf, false, bf_.bf, (short)0, acc[m][n], false, false);
    }
    __syncthreads();                   // done reading `cur` before it is refilled
  }

  // C/D layout: VGPR r -> M = half*8 + r ; N = lrow (per 16-wide subtile)
#pragma unroll
  for (int m = 0; m < 2; m++)
#pragma unroll
    for (int n = 0; n < 4; n++)
#pragma unroll
      for (int r = 0; r < 8; r++) {
        int mr = m0 + wm * 32 + m * 16 + half * 8 + r;
        int c  = n0 + wn * 64 + n * 16 + lrow;
        float val = acc[m][n][r] * alpha;
        if constexpr (OUT_BF16) Cb[(long)mr * ldc + c] = f32_to_bf16(val);
        else                    Cf[(long)mr * ldc + c] = val;
      }
}

// ---------- row softmax: f32 in, bf16 out (row length n <= 2048) ----------
__global__ __launch_bounds__(256) void softmax_rows_kernel(
    const float* __restrict__ S, unsigned short* __restrict__ P, int n) {
  const int row = blockIdx.x;
  const float* s = S + (long)row * n;
  unsigned short* p = P + (long)row * n;
  __shared__ float red[256];

  float mx = -3.0e38f;
  for (int i = threadIdx.x; i < n; i += 256) mx = fmaxf(mx, s[i]);
  red[threadIdx.x] = mx;
  __syncthreads();
  for (int off = 128; off > 0; off >>= 1) {
    if (threadIdx.x < off)
      red[threadIdx.x] = fmaxf(red[threadIdx.x], red[threadIdx.x + off]);
    __syncthreads();
  }
  mx = red[0];
  __syncthreads();

  float e[8];
  float sum = 0.0f;
  int c = 0;
  for (int i = threadIdx.x; i < n; i += 256) { e[c] = expf(s[i] - mx); sum += e[c]; c++; }
  red[threadIdx.x] = sum;
  __syncthreads();
  for (int off = 128; off > 0; off >>= 1) {
    if (threadIdx.x < off) red[threadIdx.x] += red[threadIdx.x + off];
    __syncthreads();
  }
  float inv = 1.0f / red[0];
  c = 0;
  for (int i = threadIdx.x; i < n; i += 256) { p[i] = f32_to_bf16(e[c] * inv); c++; }
}

// ---------- 4-qubit statevector circuit ----------
__device__ __forceinline__ void ry_gate(float st[16], float theta, int mask) {
  float s = sinf(0.5f * theta), c = cosf(0.5f * theta);
#pragma unroll
  for (int i = 0; i < 16; i++) {
    if ((i & mask) == 0) {
      float a0 = st[i], a1 = st[i | mask];
      st[i]        = c * a0 - s * a1;
      st[i | mask] = s * a0 + c * a1;
    }
  }
}
__device__ __forceinline__ void cnot_gate(float st[16], int cm, int tm) {
#pragma unroll
  for (int i = 0; i < 16; i++) {
    if ((i & cm) && !(i & tm)) {
      float t = st[i]; st[i] = st[i | tm]; st[i | tm] = t;
    }
  }
}

__global__ __launch_bounds__(256) void quantum_kernel(
    const float* __restrict__ attn,       // (B*T, 1024) f32
    const float* __restrict__ qparams,    // (16, 2, 4) f32
    unsigned short* __restrict__ zb) {    // (B*T, 64) bf16 packed z
  int n = blockIdx.x * 256 + threadIdx.x;   // 0 .. 131071
  int h   = n & 15;
  int row = n >> 4;

  float st[16];
#pragma unroll
  for (int i = 0; i < 16; i++) st[i] = 0.0f;
  st[0] = 1.0f;

#pragma unroll
  for (int l = 0; l < 2; l++) {
#pragma unroll
    for (int q = 0; q < 4; q++)
      ry_gate(st, qparams[h * 8 + l * 4 + q], 8 >> q);
    cnot_gate(st, 8, 4);   // (0,1)
    cnot_gate(st, 4, 2);   // (1,2)
    cnot_gate(st, 2, 1);   // (2,3)
    cnot_gate(st, 1, 8);   // (3,0)
  }
#pragma unroll
  for (int q = 0; q < 4; q++)
    ry_gate(st, attn[(long)row * 1024 + h * 64 + q], 8 >> q);

  float p[16];
#pragma unroll
  for (int i = 0; i < 16; i++) p[i] = st[i] * st[i];

#pragma unroll
  for (int q = 0; q < 4; q++) {
    int mask = 8 >> q;
    float z = 0.0f;
#pragma unroll
    for (int i = 0; i < 16; i++) z += (i & mask) ? -p[i] : p[i];
    zb[(long)row * 64 + h * 4 + q] = f32_to_bf16(z);
  }
}

// ---------- launch ----------
extern "C" void kernel_launch(void* const* d_in, const int* in_sizes, int n_in,
                              void* d_out, int out_size, void* d_ws, size_t ws_size,
                              hipStream_t stream) {
  (void)in_sizes; (void)n_in; (void)out_size; (void)ws_size;
  const float* x  = (const float*)d_in[0];
  const float* Wq = (const float*)d_in[1];
  const float* Wk = (const float*)d_in[2];
  const float* Wv = (const float*)d_in[3];
  const float* Wo = (const float*)d_in[4];
  const float* qp = (const float*)d_in[5];

  constexpr int Bn = 4, T = 2048, E = 1024, M = Bn * T;   // M = 8192

  char* ws = (char*)d_ws;
  size_t off = 0;
  auto alloc = [&](size_t bytes) -> void* {
    void* p = ws + off;
    off = (off + bytes + 255) & ~(size_t)255;
    return p;
  };
  unsigned short* xb     = (unsigned short*)alloc((size_t)M * E * 2);
  unsigned short* wqb    = (unsigned short*)alloc((size_t)E * E * 2);
  unsigned short* wkb    = (unsigned short*)alloc((size_t)E * E * 2);
  unsigned short* wvb    = (unsigned short*)alloc((size_t)E * E * 2);
  unsigned short* wob    = (unsigned short*)alloc((size_t)E * 64 * 2);
  unsigned short* qb     = (unsigned short*)alloc((size_t)M * E * 2);
  unsigned short* kb     = (unsigned short*)alloc((size_t)M * E * 2);
  unsigned short* vb     = (unsigned short*)alloc((size_t)M * E * 2);
  unsigned short* vt     = (unsigned short*)alloc((size_t)M * E * 2);   // v^T per batch
  float*          scores = (float*)         alloc((size_t)Bn * T * T * 4);
  unsigned short* probs  = (unsigned short*)alloc((size_t)Bn * T * T * 2);
  float*          attn   = (float*)         alloc((size_t)M * E * 4);
  unsigned short* zb     = (unsigned short*)alloc((size_t)M * 64 * 2);

  // 1) bf16 conversions / packing
  cvt_bf16_kernel<<<(M * E) / 1024, 256, 0, stream>>>(x,  xb,  M * E);
  cvt_bf16_kernel<<<(E * E) / 1024, 256, 0, stream>>>(Wq, wqb, E * E);
  cvt_bf16_kernel<<<(E * E) / 1024, 256, 0, stream>>>(Wk, wkb, E * E);
  cvt_bf16_kernel<<<(E * E) / 1024, 256, 0, stream>>>(Wv, wvb, E * E);
  pack_wo_kernel<<<(E * 64) / 256, 256, 0, stream>>>(Wo, wob);

  // 2) QKV projections: q/k/v = x @ W^T (bf16 out)
  dim3 gQKV(E / GBN, M / GBM, 1);
  gemm_wmma_kernel<true><<<gQKV, 256, 0, stream>>>(
      xb, wqb, qb, E, E, E, E, 0, 0, 0, 1.0f);
  gemm_wmma_kernel<true><<<gQKV, 256, 0, stream>>>(
      xb, wkb, kb, E, E, E, E, 0, 0, 0, 1.0f);
  gemm_wmma_kernel<true><<<gQKV, 256, 0, stream>>>(
      xb, wvb, vb, E, E, E, E, 0, 0, 0, 1.0f);

  // 3) scores = (q @ k^T) / sqrt(E), per batch (f32 out)
  dim3 gS(T / GBN, T / GBM, Bn);
  gemm_wmma_kernel<false><<<gS, 256, 0, stream>>>(
      qb, kb, scores, E, E, E, T,
      (long)T * E, (long)T * E, (long)T * T, 0.03125f);

  // 4) softmax rows -> bf16 probabilities
  softmax_rows_kernel<<<M, 256, 0, stream>>>(scores, probs, T);

  // 5) v^T per batch, then attn = probs @ (v^T)^T
  dim3 gT(E / 32, T / 32, Bn);
  transpose_bf16_kernel<<<gT, dim3(32, 8), 0, stream>>>(vb, vt, T, E);
  dim3 gO(E / GBN, T / GBM, Bn);
  gemm_wmma_kernel<false><<<gO, 256, 0, stream>>>(
      probs, vt, attn, T, T, T, E,
      (long)T * T, (long)T * E, (long)T * E, 1.0f);

  // 6) quantum circuit per (b,t,h) -> packed z (bf16, 64 cols)
  quantum_kernel<<<(M * 16) / 256, 256, 0, stream>>>(attn, qp, zb);

  // 7) out = zfull @ Wo^T == zpacked @ WoPacked^T (f32 out to d_out)
  dim3 gF(E / GBN, M / GBM, 1);
  gemm_wmma_kernel<false><<<gF, 256, 0, stream>>>(
      zb, wob, d_out, 64, 64, 64, E, 0, 0, 0, 1.0f);
}
